// DuelingDQN_90898687852766
// MI455X (gfx1250) — matrix operand order repaired and checked by reading, not compiled, not run on hardware
//
#include <hip/hip_runtime.h>
#include <hip/hip_bf16.h>

#define HID 128

typedef __attribute__((ext_vector_type(16))) __bf16 v16bf;
typedef __attribute__((ext_vector_type(8)))  float  v8f;

__device__ __forceinline__ __bf16 f2bf(float f) {
  union { float f; unsigned u; } v; v.f = f;
  unsigned r = (v.u + 0x7FFFu + ((v.u >> 16) & 1u)) >> 16;   // round-to-nearest-even
  union { unsigned short us; __bf16 b; } o; o.us = (unsigned short)r;
  return o.b;
}

__global__ void k_fill(float* __restrict__ p, float v, int n) {
  int t = blockIdx.x * blockDim.x + threadIdx.x;
  if (t < n) p[t] = v;
}

__global__ void k_deg(const int* __restrict__ dst, float* __restrict__ deg, int e) {
  int t = blockIdx.x * blockDim.x + threadIdx.x;
  if (t < e) atomicAdd(&deg[dst[t]], 1.0f);
}

__global__ void k_rsqrt(float* __restrict__ d, int n) {
  int t = blockIdx.x * blockDim.x + threadIdx.x;
  if (t < n) d[t] = rsqrtf(d[t]);
}

// Pack W [128x128 f32 row-major, W[k][n]] into the contiguous-k bf16 B-fragment order k_gemm_wmma uses (A loaded to match):
// index = ((colTile*4 + kb)*32 + lane)*16 + j  holds  W[kb*32 + (lane>>4)*16 + j][colTile*16 + (lane&15)]
__global__ void k_packw(const float* __restrict__ W, __bf16* __restrict__ Wp) {
  int idx = blockIdx.x * blockDim.x + threadIdx.x;
  if (idx >= 8 * 4 * 32 * 16) return;
  int j    = idx & 15;
  int lane = (idx >> 4) & 31;
  int kb   = (idx >> 9) & 3;
  int ct   = idx >> 11;
  int n = ct * 16 + (lane & 15);
  int k = kb * 32 + (lane >> 4) * 16 + j;
  Wp[idx] = f2bf(W[k * HID + n]);
}

// x [N,11] @ W1 [11,128] -> out [N,128]  (K=11 too small for WMMA; tiny GEMM, VALU is fine)
__global__ void k_gemm_in(const float* __restrict__ x, const float* __restrict__ W1,
                          float* __restrict__ out, int n) {
  __shared__ float xs[11];
  int node = blockIdx.x;
  int j = threadIdx.x;
  if (node >= n) return;
  if (j < 11) xs[j] = x[node * 11 + j];
  __syncthreads();
  float s = 0.f;
#pragma unroll
  for (int k = 0; k < 11; ++k) s += xs[k] * W1[k * HID + j];
  out[(size_t)node * HID + j] = s;
}

// One wave per edge; lane handles 4 consecutive features. agg[dst] += h[src]*dinv[src]*dinv[dst]
__global__ void k_scatter(const float* __restrict__ h, const int* __restrict__ src,
                          const int* __restrict__ dst, const float* __restrict__ dinv,
                          float* __restrict__ agg, int e) {
  int t = blockIdx.x * blockDim.x + threadIdx.x;
  int ed = t >> 5;
  if (ed >= e) return;
  int lane = t & 31;
  int s = src[ed], d = dst[ed];
  float w = dinv[s] * dinv[d];
  const float4 hv = *reinterpret_cast<const float4*>(h + (size_t)s * HID + lane * 4);
  float* ap = agg + (size_t)d * HID + lane * 4;
  atomicAdd(ap + 0, hv.x * w);
  atomicAdd(ap + 1, hv.y * w);
  atomicAdd(ap + 2, hv.z * w);
  atomicAdd(ap + 3, hv.w * w);
}

// h_out = relu(agg + hW/deg + b); also emit bf16 copy for the next WMMA GEMM
__global__ void k_combine(const float* __restrict__ agg, const float* __restrict__ hW,
                          const float* __restrict__ dinv, const float* __restrict__ bias,
                          float* __restrict__ hout, __bf16* __restrict__ hbf, int n) {
  int t = blockIdx.x * blockDim.x + threadIdx.x;
  int node = t >> 5;
  if (node >= n) return;
  int lane = t & 31;
  float di = dinv[node];
  float sw = di * di;          // 1/deg
  size_t o = (size_t)node * HID + lane * 4;
  float4 a = *reinterpret_cast<const float4*>(agg + o);
  float4 h = *reinterpret_cast<const float4*>(hW + o);
  float4 b = *reinterpret_cast<const float4*>(bias + lane * 4);
  float4 r;
  r.x = fmaxf(a.x + h.x * sw + b.x, 0.f);
  r.y = fmaxf(a.y + h.y * sw + b.y, 0.f);
  r.z = fmaxf(a.z + h.z * sw + b.z, 0.f);
  r.w = fmaxf(a.w + h.w * sw + b.w, 0.f);
  *reinterpret_cast<float4*>(hout + o) = r;
  union { __bf16 b[4]; uint2 u; } ob;
  ob.b[0] = f2bf(r.x); ob.b[1] = f2bf(r.y); ob.b[2] = f2bf(r.z); ob.b[3] = f2bf(r.w);
  *reinterpret_cast<uint2*>(hbf + o) = ob.u;
}

// A [N,128] bf16 row-major  x  Bp (packed 128x128 weights) -> C [N,128] f32.
// One wave computes one 16x16 output tile; K=128 in 4 steps of v_wmma_f32_16x16x32_bf16.
__global__ void k_gemm_wmma(const __bf16* __restrict__ A, const __bf16* __restrict__ Bp,
                            float* __restrict__ C, int n) {
  int wave = (blockIdx.x * blockDim.x + threadIdx.x) >> 5;
  int lane = threadIdx.x & 31;
  int tiles = n >> 4;
  int rowTile = wave >> 3;   // 8 column tiles of 16 cover HID=128
  int colTile = wave & 7;
  if (rowTile >= tiles) return;
  int r = lane & 15, half = lane >> 4;
  const __bf16* arow  = A + (size_t)(rowTile * 16 + r) * HID;
  const __bf16* bbase = Bp + ((size_t)colTile * 4) * 32 * 16 + (size_t)lane * 16;
  v8f acc = {};
#pragma unroll
  for (int kb = 0; kb < 4; ++kb) {
    union { v16bf v; float4 f[2]; } a, b;
    // A-fragment: lane r/half -> row r, K = kb*32 + half*16 + [0..15] (matches packed B)
    a.f[0] = *reinterpret_cast<const float4*>(arow + kb * 32 + half * 16);
    a.f[1] = *reinterpret_cast<const float4*>(arow + kb * 32 + half * 16 + 8);
    // B-fragment: pre-packed contiguous 32 bytes per lane
    const float4* bp = reinterpret_cast<const float4*>(bbase + (size_t)kb * 32 * 16);
    b.f[0] = bp[0];
    b.f[1] = bp[1];
    acc = __builtin_amdgcn_wmma_f32_16x16x32_bf16(false, a.v, false, b.v, (short)0, acc,
                                                  false, false);
  }
  // C layout: VGPR j -> row (half*8 + j), col = lane&15
  float* cp = C + (size_t)(rowTile * 16 + half * 8) * HID + colTile * 16 + r;
#pragma unroll
  for (int j = 0; j < 8; ++j) cp[(size_t)j * HID] = acc[j];
}

// Column sum of h [N,128] into g[128] (mean applied later)
__global__ void k_colsum(const float* __restrict__ h, float* __restrict__ g, int n) {
  __shared__ float s[HID];
  int t = threadIdx.x;
  if (t < HID) s[t] = 0.f;
  __syncthreads();
  size_t total = (size_t)n * HID;
  for (size_t i = (size_t)blockIdx.x * blockDim.x + threadIdx.x; i < total;
       i += (size_t)gridDim.x * blockDim.x)
    atomicAdd(&s[(int)(i & (HID - 1))], h[i]);
  __syncthreads();
  if (t < HID) atomicAdd(&g[t], s[t]);
}

// Dueling head on the pooled vector: one block of 128 threads
__global__ void k_head(const float* __restrict__ g, const float* __restrict__ Wv1,
                       const float* __restrict__ bv1, const float* __restrict__ Wv2,
                       const float* __restrict__ bv2, const float* __restrict__ Wa1,
                       const float* __restrict__ ba1, const float* __restrict__ Wa2,
                       const float* __restrict__ ba2, float* __restrict__ out, float invN) {
  __shared__ float gs[HID], hv[HID], ha[HID], red[8];
  int j = threadIdx.x;
  gs[j] = g[j] * invN;
  __syncthreads();
  float s1 = bv1[j], s2 = ba1[j];
  for (int k = 0; k < HID; ++k) {
    float gk = gs[k];
    s1 += gk * Wv1[k * HID + j];
    s2 += gk * Wa1[k * HID + j];
  }
  hv[j] = fmaxf(s1, 0.f);
  ha[j] = fmaxf(s2, 0.f);
  __syncthreads();
  if (j == 0) {
    float v = bv2[0];
    for (int k = 0; k < HID; ++k) v += hv[k] * Wv2[k];
    red[6] = v;
  }
  if (j < 6) {
    float a = ba2[j];
    for (int k = 0; k < HID; ++k) a += ha[k] * Wa2[k * 6 + j];
    red[j] = a;
  }
  __syncthreads();
  if (j == 0) {
    float m = 0.f;
    for (int a = 0; a < 6; ++a) m += red[a];
    m *= (1.0f / 6.0f);
    float v = red[6];
    for (int a = 0; a < 6; ++a) out[a] = v + red[a] - m;
  }
}

extern "C" void kernel_launch(void* const* d_in, const int* in_sizes, int n_in,
                              void* d_out, int out_size, void* d_ws, size_t ws_size,
                              hipStream_t stream) {
  (void)n_in; (void)out_size; (void)ws_size;
  const float* x   = (const float*)d_in[0];
  const int*   ei  = (const int*)d_in[1];
  const float* W1  = (const float*)d_in[2];
  const float* b1  = (const float*)d_in[3];
  const float* W2  = (const float*)d_in[4];
  const float* b2  = (const float*)d_in[5];
  const float* W3  = (const float*)d_in[6];
  const float* b3  = (const float*)d_in[7];
  const float* Wv1 = (const float*)d_in[8];
  const float* bv1 = (const float*)d_in[9];
  const float* Wv2 = (const float*)d_in[10];
  const float* bv2 = (const float*)d_in[11];
  const float* Wa1 = (const float*)d_in[12];
  const float* ba1 = (const float*)d_in[13];
  const float* Wa2 = (const float*)d_in[14];
  const float* ba2 = (const float*)d_in[15];

  const int N = in_sizes[0] / 11;
  const int E = in_sizes[1] / 2;
  const int* src = ei;
  const int* dst = ei + E;

  char* ws = (char*)d_ws;
  size_t off = 0;
  auto alloc = [&](size_t bytes) -> void* {
    void* p = ws + off;
    off += (bytes + 255) & ~(size_t)255;
    return p;
  };
  float*  dinv = (float*)alloc(sizeof(float) * (size_t)N);
  float*  bufA = (float*)alloc(sizeof(float) * (size_t)N * HID);
  float*  bufB = (float*)alloc(sizeof(float) * (size_t)N * HID);
  float*  agg  = (float*)alloc(sizeof(float) * (size_t)N * HID);
  __bf16* hbf  = (__bf16*)alloc(sizeof(__bf16) * (size_t)N * HID);
  __bf16* W2p  = (__bf16*)alloc(sizeof(__bf16) * HID * HID);
  __bf16* W3p  = (__bf16*)alloc(sizeof(__bf16) * HID * HID);
  float*  g    = (float*)alloc(sizeof(float) * HID);

  const int B = 256;
  // Degrees + normalization (computed once, reused by all three layers)
  k_fill<<<(N + B - 1) / B, B, 0, stream>>>(dinv, 1.0f, N);          // self-loop
  k_deg<<<(E + B - 1) / B, B, 0, stream>>>(dst, dinv, E);
  k_rsqrt<<<(N + B - 1) / B, B, 0, stream>>>(dinv, N);
  // Pack static weights into WMMA B-fragment layout (once per launch)
  k_packw<<<(16384 + B - 1) / B, B, 0, stream>>>(W2, W2p);
  k_packw<<<(16384 + B - 1) / B, B, 0, stream>>>(W3, W3p);
  // Layer 1 input GEMM (K=11)
  k_gemm_in<<<N, HID, 0, stream>>>(x, W1, bufA, N);

  const float*  biases[3] = {b1, b2, b3};
  const __bf16* wp[2]     = {W2p, W3p};
  const int nh = N * HID;
  const int scatterT = E * 32;
  const int combT    = N * 32;
  for (int l = 0; l < 3; ++l) {
    k_fill<<<(nh + B - 1) / B, B, 0, stream>>>(agg, 0.0f, nh);
    k_scatter<<<(scatterT + B - 1) / B, B, 0, stream>>>(bufA, src, dst, dinv, agg, E);
    k_combine<<<(combT + B - 1) / B, B, 0, stream>>>(agg, bufA, dinv, biases[l], bufB, hbf, N);
    if (l < 2) {
      int threads = (N >> 4) * 8 * 32;   // one wave per 16x16 output tile
      k_gemm_wmma<<<(threads + B - 1) / B, B, 0, stream>>>(hbf, wp[l], bufA, N);
    }
  }
  // Mean pool + dueling head
  k_fill<<<1, HID, 0, stream>>>(g, 0.0f, HID);
  k_colsum<<<1280, B, 0, stream>>>(bufB, g, N);
  k_head<<<1, HID, 0, stream>>>(g, Wv1, bv1, Wv2, bv2, Wa1, ba1, Wa2, ba2,
                                (float*)d_out, 1.0f / (float)N);
}
